// EncoderSAE_3702261809660
// MI455X (gfx1250) — matrix operand order repaired and checked by reading, not compiled, not run on hardware
//
#include <hip/hip_runtime.h>
#include <hip/hip_bf16.h>

typedef __attribute__((ext_vector_type(16))) __bf16 v16bf;
typedef __attribute__((ext_vector_type(8)))  float  v8f;

#define BATCH    2048
#define IN_DIM   1024
#define DICT     32768
#define K_SPARSE 64

// d_out offsets in floats, reference tuple order:
// reconstructed, sparse_features, l0_norm, topk_mask, raw_features
#define OFF_REC  0LL
#define OFF_SP   (2048LL * 1024LL)
#define OFF_L0   (OFF_SP + 2048LL * 32768LL)
#define OFF_MASK (OFF_L0 + 1LL)
#define OFF_RAW  (OFF_MASK + 2048LL * 32768LL)

__device__ __forceinline__ unsigned short f32_to_bf16(float f) {
  unsigned u = __float_as_uint(f);
  u += 0x7FFFu + ((u >> 16) & 1u);   // round to nearest even
  return (unsigned short)(u >> 16);
}

// ---------------------------------------------------------------- convert
__global__ void __launch_bounds__(256)
convert_bf16_kernel(const float* __restrict__ in,
                    unsigned short* __restrict__ out, long long n) {
  long long i = (long long)blockIdx.x * blockDim.x + threadIdx.x;
  long long stride = (long long)gridDim.x * blockDim.x;
  for (; i < n; i += stride) out[i] = f32_to_bf16(in[i]);
}

// ------------------------------------------------------- encoder WMMA GEMM
// features[b,d] = sum_k x[b,k] * W_enc[d,k]; writes relu->raw_features and
// zero-initializes sparse_features and topk_mask (same index space).
// Staging: GLOBAL_LOAD_ASYNC_TO_LDS_B128 with double-buffered LDS (ASYNCcnt).
__global__ void __launch_bounds__(256)
sae_encoder_wmma_kernel(const unsigned short* __restrict__ xb,   // 2048x1024 bf16
                        const unsigned short* __restrict__ wb,   // 32768x1024 bf16
                        float* __restrict__ out)                 // d_out base
{
  __shared__ __align__(32) unsigned short As[2][128][32];   // 16 KB
  __shared__ __align__(32) unsigned short Bs[2][128][32];   // 16 KB

  const int tid   = threadIdx.x;
  const int lane  = tid & 31;
  const int wave  = tid >> 5;       // 0..7
  const int waveM = wave & 3;       // 4 waves along M (32 rows each)
  const int waveN = wave >> 2;      // 2 waves along N (64 cols each)
  const int blockN = blockIdx.x;    // DICT/128 = 256
  const int blockM = blockIdx.y;    // BATCH/128 = 16
  const int l15 = lane & 15;
  const int kh  = lane >> 4;        // lane half (selects K groups per ISA layout)

  v8f zero = {};
  v8f acc[2][4];
#pragma unroll
  for (int i = 0; i < 2; ++i)
#pragma unroll
    for (int j = 0; j < 4; ++j) acc[i][j] = zero;

  const unsigned short* aG = xb + (long long)blockM * 128 * IN_DIM;
  const unsigned short* bG = wb + (long long)blockN * 128 * IN_DIM;

  // per-thread staging coords: thread covers rows {row0, row0+64}, 16B chunk ch
  const int row0 = tid >> 2;
  const int ch   = tid & 3;

  // issue one 128x32 bf16 tile (A and B) as async memory->LDS b128 transfers;
  // 4 async ops per thread => 4 ASYNCcnt increments per wave per tile
  auto issue_tile = [&](int k0, int buf) {
    const unsigned short* ga0 = aG + (long long)row0 * IN_DIM + k0 + ch * 8;
    const unsigned short* ga1 = ga0 + 64LL * IN_DIM;
    const unsigned short* gb0 = bG + (long long)row0 * IN_DIM + k0 + ch * 8;
    const unsigned short* gb1 = gb0 + 64LL * IN_DIM;
    unsigned la0 = (unsigned)(size_t)&As[buf][row0][ch * 8];
    unsigned la1 = (unsigned)(size_t)&As[buf][row0 + 64][ch * 8];
    unsigned lb0 = (unsigned)(size_t)&Bs[buf][row0][ch * 8];
    unsigned lb1 = (unsigned)(size_t)&Bs[buf][row0 + 64][ch * 8];
    asm volatile("global_load_async_to_lds_b128 %0, %1, off"
                 :: "v"(la0), "v"(ga0) : "memory");
    asm volatile("global_load_async_to_lds_b128 %0, %1, off"
                 :: "v"(la1), "v"(ga1) : "memory");
    asm volatile("global_load_async_to_lds_b128 %0, %1, off"
                 :: "v"(lb0), "v"(gb0) : "memory");
    asm volatile("global_load_async_to_lds_b128 %0, %1, off"
                 :: "v"(lb1), "v"(gb1) : "memory");
  };

  issue_tile(0, 0);

  for (int k0 = 0; k0 < IN_DIM; k0 += 32) {
    const int buf = (k0 >> 5) & 1;
    if (k0 + 32 < IN_DIM) {
      // prefetch next tile into the other buffer, then wait only for the
      // current tile's 4 ops (async ops complete in order)
      issue_tile(k0 + 32, buf ^ 1);
      asm volatile("s_wait_asynccnt 0x4" ::: "memory");
    } else {
      asm volatile("s_wait_asynccnt 0x0" ::: "memory");
    }
    __syncthreads();   // current tile visible to all waves

    const unsigned short (*A)[32] = As[buf];
    const unsigned short (*B)[32] = Bs[buf];

    // A fragments: 16-bit A 16x32 layout — lane<16: K {0..7,16..23},
    // lane>=16: K {8..15,24..31}; two 16B LDS reads per fragment.
    v16bf a[2];
#pragma unroll
    for (int tm = 0; tm < 2; ++tm) {
      int m = waveM * 32 + tm * 16 + l15;
      union { uint4 u[2]; v16bf v; } t;
      t.u[0] = *(const uint4*)&A[m][kh * 8];
      t.u[1] = *(const uint4*)&A[m][16 + kh * 8];
      a[tm] = t.v;
    }
    // B fragments: lanes 0..15 hold K {0..7,16..23}, lanes 16..31 hold K {8..15,24..31}
    v16bf bfr[4];
#pragma unroll
    for (int tn = 0; tn < 4; ++tn) {
      int n = waveN * 64 + tn * 16 + l15;
      union { uint4 u[2]; v16bf v; } t;
      t.u[0] = *(const uint4*)&B[n][kh * 8];
      t.u[1] = *(const uint4*)&B[n][16 + kh * 8];
      bfr[tn] = t.v;
    }
    // 8 back-to-back WMMAs
#pragma unroll
    for (int tn = 0; tn < 4; ++tn)
#pragma unroll
      for (int tm = 0; tm < 2; ++tm)
        acc[tm][tn] = __builtin_amdgcn_wmma_f32_16x16x32_bf16(
            false, a[tm], false, bfr[tn], (short)0, acc[tm][tn], false, false);

    __syncthreads();   // all waves done reading buf before it is re-filled
  }

  // C/D layout: VGPR r -> M = r + 8*(lane>=16); N = lane&15
  float* raw  = out + OFF_RAW;
  float* sp   = out + OFF_SP;
  float* mask = out + OFF_MASK;
#pragma unroll
  for (int tm = 0; tm < 2; ++tm) {
#pragma unroll
    for (int tn = 0; tn < 4; ++tn) {
      int gcol = blockN * 128 + waveN * 64 + tn * 16 + l15;
#pragma unroll
      for (int r = 0; r < 8; ++r) {
        int grow = blockM * 128 + waveM * 32 + tm * 16 + r + (kh << 3);
        long long o = (long long)grow * DICT + gcol;
        float v = acc[tm][tn][r];
        raw[o]  = v > 0.0f ? v : 0.0f;
        sp[o]   = 0.0f;
        mask[o] = 0.0f;
      }
    }
  }
}

// ---------------------------------------------------------------- l0 init
__global__ void zero_l0_kernel(float* out) {
  if (threadIdx.x == 0 && blockIdx.x == 0) out[OFF_L0] = 0.0f;
}

// ----------------------------------------------------- top-k radix select
// One block per row. Values are >=0 (post-ReLU) so float bit pattern order
// equals numeric order. Two 8-bit radix passes give a 16-bit threshold
// prefix; final pass gathers with a capped atomic counter (ties arbitrary).
__global__ void __launch_bounds__(256)
topk_kernel(float* __restrict__ out, float* __restrict__ tv, int* __restrict__ ti) {
  const int b   = blockIdx.x;
  const int tid = threadIdx.x;
  const float* row = out + OFF_RAW + (long long)b * DICT;

  __shared__ int hist[256];
  __shared__ int sSel, sRem, sSel2, sRem2;
  __shared__ float vals[K_SPARSE];
  __shared__ int   idxs[K_SPARSE];
  __shared__ int cnt, cntEq, posCnt;

  // pass 1: histogram on bits[30:23]
  hist[tid] = 0;
  __syncthreads();
  for (int i = tid; i < DICT; i += 256) {
    unsigned u = __float_as_uint(row[i]);
    atomicAdd(&hist[u >> 23], 1);
  }
  __syncthreads();
  if (tid == 0) {
    int above = 0, bin;
    for (bin = 255; bin >= 0; --bin) {
      int c = hist[bin];
      if (above + c >= K_SPARSE || bin == 0) break;
      above += c;
    }
    sSel = bin;
    sRem = K_SPARSE - above;
  }
  __syncthreads();
  const int selBin = sSel, rem = sRem;

  // pass 2: histogram on bits[22:15] within selected exponent bin
  hist[tid] = 0;
  __syncthreads();
  for (int i = tid; i < DICT; i += 256) {
    unsigned u = __float_as_uint(row[i]);
    if ((int)(u >> 23) == selBin) atomicAdd(&hist[(u >> 15) & 0xFF], 1);
  }
  __syncthreads();
  if (tid == 0) {
    int above = 0, bin;
    for (bin = 255; bin >= 0; --bin) {
      int c = hist[bin];
      if (above + c >= rem || bin == 0) break;
      above += c;
    }
    sSel2 = bin;
    sRem2 = rem - above;
    cnt = 0; cntEq = 0; posCnt = 0;
  }
  __syncthreads();
  const unsigned thr  = ((unsigned)selBin << 8) | (unsigned)sSel2;  // on bits>>15
  const int      rem2 = sRem2;

  // pass 3: gather
  for (int i = tid; i < DICT; i += 256) {
    float v = row[i];
    unsigned p = __float_as_uint(v) >> 15;
    if (p > thr) {
      int s = atomicAdd(&cnt, 1);
      if (s < K_SPARSE) { vals[s] = v; idxs[s] = i; }
    } else if (p == thr) {
      int e = atomicAdd(&cntEq, 1);
      if (e < rem2) {
        int s = atomicAdd(&cnt, 1);
        if (s < K_SPARSE) { vals[s] = v; idxs[s] = i; }
      }
    }
  }
  __syncthreads();
  const int total = cnt < K_SPARSE ? cnt : K_SPARSE;

  if (tid < K_SPARSE) {
    float v = 0.0f; int id = -1;
    if (tid < total) { v = vals[tid]; id = idxs[tid]; }
    tv[b * K_SPARSE + tid] = v;
    ti[b * K_SPARSE + tid] = id;
    if (id >= 0) {
      out[OFF_SP   + (long long)b * DICT + id] = v;
      out[OFF_MASK + (long long)b * DICT + id] = 1.0f;
      if (v > 0.0f) atomicAdd(&posCnt, 1);
    }
  }
  __syncthreads();
  if (tid == 0) atomicAdd(&out[OFF_L0], (float)posCnt * (1.0f / BATCH));
}

// ----------------------------------------------- W_dec transpose (1024x32768 -> 32768x1024)
__global__ void __launch_bounds__(256)
transpose_kernel(const float* __restrict__ in, float* __restrict__ outT) {
  __shared__ float tile[32][33];
  int x  = blockIdx.x * 32 + threadIdx.x;   // dict dim
  int y0 = blockIdx.y * 32;                 // input dim
  for (int j = threadIdx.y; j < 32; j += 8)
    tile[j][threadIdx.x] = in[(long long)(y0 + j) * DICT + x];
  __syncthreads();
  int xo  = blockIdx.y * 32 + threadIdx.x;  // input dim
  int yo0 = blockIdx.x * 32;                // dict dim
  for (int j = threadIdx.y; j < 32; j += 8)
    outT[(long long)(yo0 + j) * IN_DIM + xo] = tile[threadIdx.x][j];
}

// --------------------------------------------------------- sparse decoder
// reconstructed[b,i] = sum_j val[b][j] * W_decT[idx[b][j], i]
__global__ void __launch_bounds__(256)
decoder_kernel(const float* __restrict__ wdt, const float* __restrict__ tv,
               const int* __restrict__ ti, float* __restrict__ rec) {
  const int b = blockIdx.x;
  const int t = threadIdx.x;
  __shared__ float sv[K_SPARSE];
  __shared__ int   si[K_SPARSE];
  if (t < K_SPARSE) {
    sv[t] = tv[b * K_SPARSE + t];
    int id = ti[b * K_SPARSE + t];
    si[t] = id < 0 ? 0 : id;       // value is 0 when id invalid
  }
  __syncthreads();
  float a0 = 0.f, a1 = 0.f, a2 = 0.f, a3 = 0.f;
  for (int j = 0; j < K_SPARSE; ++j) {
    float v = sv[j];
    const float* wr = wdt + (long long)si[j] * IN_DIM;
    a0 += v * wr[t];
    a1 += v * wr[t + 256];
    a2 += v * wr[t + 512];
    a3 += v * wr[t + 768];
  }
  long long o = (long long)b * IN_DIM + t;
  rec[o]       = a0;
  rec[o + 256] = a1;
  rec[o + 512] = a2;
  rec[o + 768] = a3;
}

// ---------------------------------------------------------------- launch
extern "C" void kernel_launch(void* const* d_in, const int* in_sizes, int n_in,
                              void* d_out, int out_size, void* d_ws, size_t ws_size,
                              hipStream_t stream) {
  const float* x     = (const float*)d_in[0];
  const float* W_enc = (const float*)d_in[1];
  const float* W_dec = (const float*)d_in[2];
  float* out = (float*)d_out;

  char* ws = (char*)d_ws;
  unsigned short* xb  = (unsigned short*)(ws);                                  //   4 MB
  unsigned short* wb  = (unsigned short*)(ws + (4LL << 20));                    //  64 MB
  float*          wdt = (float*)(ws + (68LL << 20));                            // 128 MB
  float*          tv  = (float*)(ws + (196LL << 20));                           // 512 KB
  int*            ti  = (int*)(ws + (196LL << 20) + 2048LL * K_SPARSE * 4);     // 512 KB

  convert_bf16_kernel<<<2048, 256, 0, stream>>>(x, xb, 2048LL * 1024);
  convert_bf16_kernel<<<8192, 256, 0, stream>>>(W_enc, wb, 32768LL * 1024);

  dim3 ggrid(DICT / 128, BATCH / 128);
  sae_encoder_wmma_kernel<<<ggrid, 256, 0, stream>>>(xb, wb, out);

  zero_l0_kernel<<<1, 32, 0, stream>>>(out);
  topk_kernel<<<BATCH, 256, 0, stream>>>(out, tv, ti);

  dim3 tgrid(DICT / 32, IN_DIM / 32);
  transpose_kernel<<<tgrid, dim3(32, 8), 0, stream>>>(W_dec, wdt);
  decoder_kernel<<<BATCH, 256, 0, stream>>>(wdt, tv, ti, out + OFF_REC);
}
